// DensityDenoisingLayer_30580167147698
// MI455X (gfx1250) — matrix operand order repaired and checked by reading, not compiled, not run on hardware
//
#include <hip/hip_runtime.h>
#include <hip/hip_bf16.h>

// ---------------------------------------------------------------------------
// Edge-MLP + LayerNorm + residual for MI455X (gfx1250), bf16 WMMA path.
//
// h  = ReLU([src_inv|dst_inv|ef] @ W1 + b1) @ W2 + b2     (bf16 A/B, f32 acc)
// out = ef + LayerNorm(h) * gamma + beta                   (f32)
//
// One wave32 processes a 16-edge tile:
//   L1: 16 accum tiles (N=256) x 12 K-steps of v_wmma_f32_16x16x32_bf16
//   L2: 16 accum tiles x 8 K-steps, A staged through LDS (row-major bf16)
//   LN: LDS f32 tile, per-lane half-row reduce + shfl_xor(16) combine
// ---------------------------------------------------------------------------

typedef __attribute__((ext_vector_type(16))) __bf16 v16bf;
typedef __attribute__((ext_vector_type(8)))  __bf16 v8bf;
typedef __attribute__((ext_vector_type(8)))  float  v8f;
typedef __attribute__((ext_vector_type(4)))  float  v4f;

#define H_DIM   256
#define INV_DIM 64
#define IN_DIM  384
#define TILE_M  16
#define WAVES   4
#define LDS_PER_WAVE 16384   // max(16x256 bf16 = 8KB, 16x256 f32 = 16KB)

// ---------------------------------------------------------------------------
// Pass 1: W1[384x256], W2[256x256] (f32, [in,out]) -> bf16 transposed [out,in]
// so every WMMA B-fragment is a single contiguous 32B run per lane.
// ---------------------------------------------------------------------------
__global__ void prep_weights_kernel(const float* __restrict__ W1,
                                    const float* __restrict__ W2,
                                    __bf16* __restrict__ W1t,
                                    __bf16* __restrict__ W2t) {
    int idx = blockIdx.x * blockDim.x + threadIdx.x;
    if (idx < IN_DIM * H_DIM) {                 // W1: k = idx/256, n = idx%256
        int k = idx >> 8, n = idx & 255;
        W1t[n * IN_DIM + k] = (__bf16)W1[idx];
    }
    if (idx < H_DIM * H_DIM) {                  // W2
        int k = idx >> 8, n = idx & 255;
        W2t[n * H_DIM + k] = (__bf16)W2[idx];
    }
}

// ---------------------------------------------------------------------------
// Main fused kernel.
// ---------------------------------------------------------------------------
__global__ __launch_bounds__(WAVES * 32)
void edge_mlp_ln_kernel(const float* __restrict__ node_inv,   // [N,64]
                        const float* __restrict__ ef,          // [E,256]
                        const int*   __restrict__ edge_index,  // [2,E]
                        const float* __restrict__ b1,
                        const float* __restrict__ b2,
                        const float* __restrict__ gamma,
                        const float* __restrict__ beta,
                        const __bf16* __restrict__ W1t,        // [256,384] bf16
                        const __bf16* __restrict__ W2t,        // [256,256] bf16
                        float* __restrict__ out,               // [E,256]
                        int n_edges) {
    __shared__ __align__(16) char smem_raw[WAVES * LDS_PER_WAVE];

    const int lane = threadIdx.x & 31;
    const int wave = threadIdx.x >> 5;
    char* my = smem_raw + wave * LDS_PER_WAVE;
    __bf16* Hs = (__bf16*)my;   // [16][256] bf16 hidden tile (layer-1 out)
    float*  Cs = (float*)my;    // [16][256] f32 pre-LN tile (overlays Hs)

    const int tile = blockIdx.x * WAVES + wave;
    const long row0 = (long)tile * TILE_M;
    if (row0 + TILE_M > n_edges) return;        // wave-uniform guard (EXEC all-1)

    const int mrow = lane & 15;                 // matrix row (A) / col (B,C)
    const int half = lane >> 4;                 // lane-group selector
    const long row = row0 + mrow;               // this lane's edge row

    const int src = edge_index[row];
    const int dst = edge_index[(long)n_edges + row];
    const float* srcp = node_inv + (long)src * INV_DIM;
    const float* dstp = node_inv + (long)dst * INV_DIM;
    const float* efp  = ef + row * (long)H_DIM;

    // ------------------------- Layer 1: X[16x384] @ W1 ---------------------
    v8f acc[16];
    const v8f vzero = {0.f, 0.f, 0.f, 0.f, 0.f, 0.f, 0.f, 0.f};
#pragma unroll
    for (int j = 0; j < 16; ++j) acc[j] = vzero;

    for (int k = 0; k < IN_DIM / 32; ++k) {
        const int kb = k * 32;
        // select the concat source for this K-tile (boundaries align to 32)
        const float* xs; int off;
        if (kb < INV_DIM)          { xs = srcp; off = kb; }
        else if (kb < 2 * INV_DIM) { xs = dstp; off = kb - INV_DIM; }
        else                       { xs = efp;  off = kb - 2 * INV_DIM; }
        // A fragment: lane holds K-run [half*16,+16)
        const float* p0 = xs + off + half * 16;
        const float* p1 = xs + off + half * 16 + 8;
        v4f x0 = *(const v4f*)(p0);
        v4f x1 = *(const v4f*)(p0 + 4);
        v4f x2 = *(const v4f*)(p1);
        v4f x3 = *(const v4f*)(p1 + 4);
        v16bf a;
#pragma unroll
        for (int t = 0; t < 4; ++t) {
            a[t]      = (__bf16)x0[t];
            a[4 + t]  = (__bf16)x1[t];
            a[8 + t]  = (__bf16)x2[t];
            a[12 + t] = (__bf16)x3[t];
        }
#pragma unroll
        for (int j = 0; j < 16; ++j) {
            // B fragment: contiguous 16 bf16 at W1t[n][kb + half*16]
            const __bf16* w = W1t + (j * 16 + mrow) * IN_DIM + kb + half * 16;
            v16bf b = *(const v16bf*)w;
            acc[j] = __builtin_amdgcn_wmma_f32_16x16x32_bf16(
                false, a, false, b, (short)0, acc[j], false, false);
        }
    }

    // bias + ReLU -> bf16 hidden tile in LDS (row-major [m][n])
#pragma unroll
    for (int j = 0; j < 16; ++j) {
        const int n = j * 16 + mrow;
        const float bias = b1[n];
#pragma unroll
        for (int r = 0; r < 8; ++r) {
            float v = acc[j][r] + bias;
            v = v > 0.f ? v : 0.f;
            Hs[(half * 8 + r) * H_DIM + n] = (__bf16)v;
        }
    }

    // ------------------------- Layer 2: H[16x256] @ W2 ---------------------
    v8f acc2[16];
#pragma unroll
    for (int j = 0; j < 16; ++j) acc2[j] = vzero;

    for (int k = 0; k < H_DIM / 32; ++k) {
        const int kb = k * 32;
        const __bf16* hp = Hs + mrow * H_DIM + kb + half * 16;
        v8bf h0 = *(const v8bf*)(hp);        // ds_load_b128
        v8bf h1 = *(const v8bf*)(hp + 8);    // ds_load_b128
        v16bf a;
#pragma unroll
        for (int t = 0; t < 8; ++t) { a[t] = h0[t]; a[8 + t] = h1[t]; }
#pragma unroll
        for (int j = 0; j < 16; ++j) {
            const __bf16* w = W2t + (j * 16 + mrow) * H_DIM + kb + half * 16;
            v16bf b = *(const v16bf*)w;
            acc2[j] = __builtin_amdgcn_wmma_f32_16x16x32_bf16(
                false, a, false, b, (short)0, acc2[j], false, false);
        }
    }

    // bias -> f32 pre-LN tile in LDS (overlays Hs; same-wave DS ops in order)
#pragma unroll
    for (int j = 0; j < 16; ++j) {
        const int n = j * 16 + mrow;
        const float bias = b2[n];
#pragma unroll
        for (int r = 0; r < 8; ++r)
            Cs[(half * 8 + r) * H_DIM + n] = acc2[j][r] + bias;
    }

    // ------------------- LayerNorm + residual + store ----------------------
    // lane handles row mrow, columns [half*128, half*128+128)
    const int cbase = half * 128;
    const float* crow = Cs + mrow * H_DIM + cbase;
    float s = 0.f, sq = 0.f;
#pragma unroll
    for (int i = 0; i < 32; ++i) {
        v4f v = *(const v4f*)(crow + i * 4);
        s  += v[0] + v[1] + v[2] + v[3];
        sq += v[0] * v[0] + v[1] * v[1] + v[2] * v[2] + v[3] * v[3];
    }
    s  += __shfl_xor(s, 16, 32);    // combine with partner half-row (wave32)
    sq += __shfl_xor(sq, 16, 32);
    const float mu  = s * (1.f / H_DIM);
    const float var = sq * (1.f / H_DIM) - mu * mu;
    const float rs  = rsqrtf(var + 1e-5f);

    const float* efrow = efp + cbase;
    float* orow = out + row * (long)H_DIM + cbase;
#pragma unroll
    for (int i = 0; i < 32; ++i) {
        v4f v = *(const v4f*)(crow + i * 4);
        v4f g = *(const v4f*)(gamma + cbase + i * 4);
        v4f bt = *(const v4f*)(beta + cbase + i * 4);
        v4f e = *(const v4f*)(efrow + i * 4);
        v4f o;
#pragma unroll
        for (int t = 0; t < 4; ++t)
            o[t] = (v[t] - mu) * rs * g[t] + bt[t] + e[t];
        *(v4f*)(orow + i * 4) = o;
    }
}

// ---------------------------------------------------------------------------
extern "C" void kernel_launch(void* const* d_in, const int* in_sizes, int n_in,
                              void* d_out, int out_size, void* d_ws, size_t ws_size,
                              hipStream_t stream) {
    const float* node_inv = (const float*)d_in[0];
    const float* ef       = (const float*)d_in[1];
    const int*   eidx     = (const int*)d_in[2];
    const float* W1       = (const float*)d_in[3];
    const float* b1       = (const float*)d_in[4];
    const float* W2       = (const float*)d_in[5];
    const float* b2       = (const float*)d_in[6];
    const float* gamma    = (const float*)d_in[7];
    const float* beta     = (const float*)d_in[8];
    float* out = (float*)d_out;

    const int n_edges = in_sizes[1] / H_DIM;   // edge_features flat / 256

    // workspace: bf16 transposed weights
    __bf16* W1t = (__bf16*)d_ws;                                   // 256*384*2
    __bf16* W2t = (__bf16*)((char*)d_ws + (size_t)H_DIM * IN_DIM * 2);

    {
        int total = IN_DIM * H_DIM;            // covers both (W2 smaller)
        int blocks = (total + 255) / 256;
        prep_weights_kernel<<<blocks, 256, 0, stream>>>(W1, W2, W1t, W2t);
    }
    {
        int tiles  = (n_edges + TILE_M - 1) / TILE_M;   // 50000 (exact)
        int blocks = (tiles + WAVES - 1) / WAVES;       // 12500
        edge_mlp_ln_kernel<<<blocks, WAVES * 32, 0, stream>>>(
            node_inv, ef, eidx, b1, b2, gamma, beta, W1t, W2t, out, n_edges);
    }
}
